// Attention_23974507446850
// MI455X (gfx1250) — matrix operand order repaired and checked by reading, not compiled, not run on hardware
//
#include <hip/hip_runtime.h>
#include <hip/hip_bf16.h>

// ---------------------------------------------------------------------------
// Fused attention block for MI455X (gfx1250, wave32, WMMA).
// Everything heavy runs on v_wmma_f32_16x16x32_bf16 with fp32 accumulation.
//
// Fragment-layout assumptions (CDNA5 ISA ch 7.12.2):
//  A (16x32 bf16): lane holds row m=lane%16, K elems {8h+0..7, 8h+16..23}, h=lane/16
//  B (32x16 bf16): lane holds col n=lane%16, K elems {8h+0..7, 8h+16..23} (attn: A and B both use 16h+0..15)
//  C/D (16x16 f32): lane holds n=lane%16, m = vgpr + 8h
//
// Wave-uniform control flow goes through readfirstlane so tile-skipping
// compiles to scalar branches (EXEC stays all-ones around every WMMA).
// ---------------------------------------------------------------------------

#define HEADS    16
#define DIM_HEAD 64
#define INNER    1024
#define QDIM     1024
#define CDIM     1024
#define BATCH    4
#define SEQ      2048

typedef __bf16 bf16;
typedef __attribute__((ext_vector_type(16))) bf16  v16bf;
typedef __attribute__((ext_vector_type(8)))  bf16  bf16x8;
typedef __attribute__((ext_vector_type(4)))  bf16  bf16x4;
typedef __attribute__((ext_vector_type(2)))  bf16  bf16x2;
typedef __attribute__((ext_vector_type(8)))  float v8f;
typedef __attribute__((ext_vector_type(4)))  float f32x4;

union FragAB { v16bf v; bf16x8 h[2]; };

static __device__ inline bf16 f2bf(float f) {
  unsigned u = __builtin_bit_cast(unsigned, f);
  u += 0x7FFFu + ((u >> 16) & 1u);                 // round to nearest even
  unsigned short hs = (unsigned short)(u >> 16);
  return __builtin_bit_cast(bf16, hs);
}
static __device__ inline float bf2f(bf16 b) {
  unsigned short hs = __builtin_bit_cast(unsigned short, b);
  unsigned u = ((unsigned)hs) << 16;
  return __builtin_bit_cast(float, u);
}

// ---------------------------------------------------------------------------
// Context projections: k0/v0 = context @ Wk/Wv  (tiny: 2*4*1024 outputs)
// ---------------------------------------------------------------------------
__global__ __launch_bounds__(256) void ctxproj_kernel(
    const float* __restrict__ ctx, const float* __restrict__ Wk,
    const float* __restrict__ Wv, bf16* __restrict__ k0, bf16* __restrict__ v0)
{
  int gid   = blockIdx.x * 256 + threadIdx.x;     // 0..8191
  int which = gid >> 12;                          // 0 = k, 1 = v
  int b     = (gid >> 10) & 3;
  int col   = gid & 1023;
  const float* Wm = which ? Wv : Wk;
  float acc = 0.0f;
  for (int k = 0; k < CDIM; ++k)
    acc += ctx[b * CDIM + k] * Wm[(size_t)k * INNER + col];
  (which ? v0 : k0)[b * INNER + col] = f2bf(acc);
}

// ---------------------------------------------------------------------------
// WMMA GEMM: C[M,Nc] = A[M,K] @ W[K,Nc].  Block tile 128x128, K-step 32.
// 8 waves: 2 (M) x 4 (N); each wave -> 64x32 via 4x2 grid of 16x16 WMMAs.
// ---------------------------------------------------------------------------
template <bool A_BF16, bool OUT_BF16>
__global__ __launch_bounds__(256) void gemm128_kernel(
    const void* __restrict__ Ap, const float* __restrict__ W,
    void* __restrict__ Cp, const float* __restrict__ bias,
    int M, int K, int Nc)
{
  __shared__ alignas(16) bf16 Asm[128][40];   // [m][k] k-contig, pad 32->40
  __shared__ alignas(16) bf16 Bsm[128][40];   // [n][k] transposed, k-contig

  const int tid = threadIdx.x;
  const int lane = tid & 31, w = tid >> 5;
  const int lr = lane & 15, hh = lane >> 4;
  const int wm = w & 1, wn = w >> 1;
  const int blockN = blockIdx.x * 128;
  const int blockM = blockIdx.y * 128;

  v8f acc[4][2];
#pragma unroll
  for (int i = 0; i < 4; ++i)
#pragma unroll
    for (int j = 0; j < 2; ++j) acc[i][j] = (v8f){0,0,0,0,0,0,0,0};

  for (int k0 = 0; k0 < K; k0 += 32) {
    // --- stage A (128x32) and B (32x128, transposed) into LDS ---
#pragma unroll
    for (int r = 0; r < 4; ++r) {
      int idx = tid + 256 * r;
      int m  = idx >> 3;
      int k4 = (idx & 7) * 4;
      if (A_BF16) {
        const bf16* A = (const bf16*)Ap;
        bf16x4 vv = *(const bf16x4*)(A + (size_t)(blockM + m) * K + k0 + k4);
        *(bf16x4*)&Asm[m][k4] = vv;
      } else {
        const float* A = (const float*)Ap;
        f32x4 vv = *(const f32x4*)(A + (size_t)(blockM + m) * K + k0 + k4);
        bf16x4 o; o[0]=f2bf(vv[0]); o[1]=f2bf(vv[1]); o[2]=f2bf(vv[2]); o[3]=f2bf(vv[3]);
        *(bf16x4*)&Asm[m][k4] = o;
      }
      int kk = idx >> 5;
      int n4 = (idx & 31) * 4;
      f32x4 wv = *(const f32x4*)(W + (size_t)(k0 + kk) * Nc + blockN + n4);
#pragma unroll
      for (int e = 0; e < 4; ++e) Bsm[n4 + e][kk] = f2bf(wv[e]);
    }
    __syncthreads();

    FragAB af[4], bfq[2];
#pragma unroll
    for (int tm = 0; tm < 4; ++tm) {
      int row = wm * 64 + tm * 16 + lr;
      af[tm].h[0] = *(const bf16x8*)&Asm[row][8 * hh];
      af[tm].h[1] = *(const bf16x8*)&Asm[row][8 * hh + 16];
    }
#pragma unroll
    for (int tn = 0; tn < 2; ++tn) {
      int col = wn * 32 + tn * 16 + lr;
      bfq[tn].h[0] = *(const bf16x8*)&Bsm[col][8 * hh];
      bfq[tn].h[1] = *(const bf16x8*)&Bsm[col][8 * hh + 16];
    }
#pragma unroll
    for (int tm = 0; tm < 4; ++tm)
#pragma unroll
      for (int tn = 0; tn < 2; ++tn)
        acc[tm][tn] = __builtin_amdgcn_wmma_f32_16x16x32_bf16(
            false, af[tm].v, false, bfq[tn].v, (short)0, acc[tm][tn], false, false);
    __syncthreads();
  }

  // --- epilogue ---
#pragma unroll
  for (int tm = 0; tm < 4; ++tm)
#pragma unroll
    for (int tn = 0; tn < 2; ++tn)
#pragma unroll
      for (int e = 0; e < 8; ++e) {
        int gm = blockM + wm * 64 + tm * 16 + 8 * hh + e;
        int gn = blockN + wn * 32 + tn * 16 + lr;
        if (OUT_BF16)
          ((bf16*)Cp)[(size_t)gm * Nc + gn] = f2bf(acc[tm][tn][e]);
        else
          ((float*)Cp)[(size_t)gm * Nc + gn] = acc[tm][tn][e] + bias[gn];
      }
}

// ---------------------------------------------------------------------------
// Flash attention: keys/values = [ctx-token ; Q].  Computes S^T = K*Q^T so the
// softmax'd P converts to the PV B-fragment with one shfl_xor(16) exchange.
// Block = 128 queries of one (b,h); wave = 16 queries. Output O^T in regs.
// ---------------------------------------------------------------------------
__global__ __launch_bounds__(256) void attn128_kernel(
    const bf16* __restrict__ Qb, const bf16* __restrict__ K0,
    const bf16* __restrict__ V0, bf16* __restrict__ AO)
{
  __shared__ alignas(16) bf16 Kt[128][72];    // [j][d], pad 64->72
  __shared__ alignas(16) bf16 Vt[64][136];    // [d][j], pad 128->136

  const int tid = threadIdx.x;
  const int lane = tid & 31;
  // wave id: wave-uniform -> force into an SGPR so tile guards become scalar
  // branches and EXEC stays all-ones around every WMMA.
  const int w  = __builtin_amdgcn_readfirstlane(tid >> 5);
  const int lr = lane & 15, hh = lane >> 4;
  const int qt = blockIdx.x, h = blockIdx.y, b = blockIdx.z;
  const int q16 = qt * 128 + w * 16;

  // Q fragment (B-frag of Q^T): lane holds q=lr, d = 32*dh + 16*hh + e.
  // Pre-scale by 1/sqrt(64)=0.125 (exact exponent shift in bf16).
  const bf16* qrow = Qb + (size_t)(b * SEQ + q16 + lr) * INNER + h * DIM_HEAD;
  FragAB qf0, qf1;
  {
    bf16x8 a = *(const bf16x8*)(qrow + 16 * hh);
    bf16x8 c = *(const bf16x8*)(qrow + 16 * hh + 8);
    bf16x8 d = *(const bf16x8*)(qrow + 32 + 16 * hh);
    bf16x8 g = *(const bf16x8*)(qrow + 32 + 16 * hh + 8);
#pragma unroll
    for (int e = 0; e < 8; ++e) {
      qf0.h[0][e] = f2bf(bf2f(a[e]) * 0.125f);
      qf0.h[1][e] = f2bf(bf2f(c[e]) * 0.125f);
      qf1.h[0][e] = f2bf(bf2f(d[e]) * 0.125f);
      qf1.h[1][e] = f2bf(bf2f(g[e]) * 0.125f);
    }
  }

  // Context score: s_ctx = (q*scale) . k0   (lane halves hold complementary d)
  const bf16* k0r = K0 + b * INNER + h * DIM_HEAD;
  float sctx = 0.0f;
  {
    bf16x8 a = *(const bf16x8*)(k0r + 16 * hh);
    bf16x8 c = *(const bf16x8*)(k0r + 16 * hh + 8);
    bf16x8 d = *(const bf16x8*)(k0r + 32 + 16 * hh);
    bf16x8 g = *(const bf16x8*)(k0r + 32 + 16 * hh + 8);
#pragma unroll
    for (int e = 0; e < 8; ++e)
      sctx += bf2f(qf0.h[0][e]) * bf2f(a[e]) + bf2f(qf0.h[1][e]) * bf2f(c[e]) +
              bf2f(qf1.h[0][e]) * bf2f(d[e]) + bf2f(qf1.h[1][e]) * bf2f(g[e]);
  }
  sctx += __shfl_xor(sctx, 16, 32);

  // v0 in O^T accumulator layout: o[td][e] lives at d = 16*td + 8*hh + e
  const bf16* v0r = V0 + b * INNER + h * DIM_HEAD;
  bf16x8 vv[4];
#pragma unroll
  for (int td = 0; td < 4; ++td)
    vv[td] = *(const bf16x8*)(v0r + 16 * td + 8 * hh);

  // Online softmax init with the (always unmasked) context token.
  float m_run = sctx, l_run = 1.0f;
  v8f o[4];
#pragma unroll
  for (int td = 0; td < 4; ++td)
#pragma unroll
    for (int e = 0; e < 8; ++e) o[td][e] = bf2f(vv[td][e]);

  for (int kt = 0; kt <= qt; ++kt) {
    // --- stage 128 key rows: Kt[j][d] (b128) and transposed Vt[d][j].
    //     Each thread stages 2 adjacent rows so the transpose packs pairs
    //     (j, j+1) into single 4-byte LDS stores. ---
    {
      int j0  = (tid & 63) * 2;     // even key row
      int seg = tid >> 6;           // d-range: seg*16 .. seg*16+15
      const bf16* src0 = Qb + (size_t)(b * SEQ + kt * 128 + j0) * INNER +
                         h * DIM_HEAD + seg * 16;
      const bf16* src1 = src0 + INNER;
      bf16x8 a0 = *(const bf16x8*)(src0);
      bf16x8 a1 = *(const bf16x8*)(src0 + 8);
      bf16x8 b0 = *(const bf16x8*)(src1);
      bf16x8 b1 = *(const bf16x8*)(src1 + 8);
      *(bf16x8*)&Kt[j0][seg * 16]         = a0;
      *(bf16x8*)&Kt[j0][seg * 16 + 8]     = a1;
      *(bf16x8*)&Kt[j0 + 1][seg * 16]     = b0;
      *(bf16x8*)&Kt[j0 + 1][seg * 16 + 8] = b1;
#pragma unroll
      for (int e = 0; e < 8; ++e) {
        bf16x2 p0; p0[0] = a0[e]; p0[1] = b0[e];
        *(bf16x2*)&Vt[seg * 16 + e][j0] = p0;
        bf16x2 p1; p1[0] = a1[e]; p1[1] = b1[e];
        *(bf16x2*)&Vt[seg * 16 + 8 + e][j0] = p1;
      }
    }
    __syncthreads();

    // jtmax is wave-uniform (scalar): fully-masked subtiles skipped with
    // scalar branches, never with EXEC masking.
    const int diag  = (kt == qt);
    const int jtmax = diag ? w : 7;
    v8f st[8];
#pragma unroll
    for (int jt = 0; jt < 8; ++jt) if (jt <= jtmax) {
      v8f a = (v8f){0,0,0,0,0,0,0,0};
      int row = jt * 16 + lr;
      FragAB kf;
      kf.h[0] = *(const bf16x8*)&Kt[row][16 * hh];
      kf.h[1] = *(const bf16x8*)&Kt[row][16 * hh + 8];
      a = __builtin_amdgcn_wmma_f32_16x16x32_bf16(false, kf.v, false, qf0.v,
                                                  (short)0, a, false, false);
      kf.h[0] = *(const bf16x8*)&Kt[row][32 + 16 * hh];
      kf.h[1] = *(const bf16x8*)&Kt[row][32 + 16 * hh + 8];
      a = __builtin_amdgcn_wmma_f32_16x16x32_bf16(false, kf.v, false, qf1.v,
                                                  (short)0, a, false, false);
      if (diag && jt == jtmax) {            // diagonal 16x16: key<=query
#pragma unroll
        for (int e = 0; e < 8; ++e)
          a[e] = (e + 8 * hh > lr) ? -3.0e38f : a[e];
      }
      st[jt] = a;
    }

    // --- online softmax update (lane pair q / q+16 holds key halves) ---
    float tmax = -3.0e38f;
#pragma unroll
    for (int jt = 0; jt < 8; ++jt) if (jt <= jtmax)
#pragma unroll
      for (int e = 0; e < 8; ++e) tmax = fmaxf(tmax, st[jt][e]);
    tmax = fmaxf(tmax, __shfl_xor(tmax, 16, 32));
    float m_new = fmaxf(m_run, tmax);
    float alpha = __expf(m_run - m_new);
    float rsum = 0.0f;
#pragma unroll
    for (int jt = 0; jt < 8; ++jt) if (jt <= jtmax)
#pragma unroll
      for (int e = 0; e < 8; ++e) {
        float p = __expf(st[jt][e] - m_new);
        st[jt][e] = p;
        rsum += p;
      }
    rsum += __shfl_xor(rsum, 16, 32);
    l_run = l_run * alpha + rsum;
    m_run = m_new;
#pragma unroll
    for (int td = 0; td < 4; ++td) o[td] *= alpha;

    // --- O^T += V^T * P^T : P^T B-frag built from st via one lane-pair swap ---
#pragma unroll
    for (int ks = 0; ks < 4; ++ks) if (2 * ks <= jtmax) {
      const bool has_hi = (2 * ks + 1 <= jtmax);
      FragAB pf;
#pragma unroll
      for (int e = 0; e < 8; ++e) {
        float own2k  = st[2 * ks][e];
        float own2k1 = has_hi ? st[2 * ks + 1][e] : 0.0f;
        float send = hh ? own2k : own2k1;
        float recv = __shfl_xor(send, 16, 32);
        float lo = hh ? recv : own2k;      // k element e    (=16*hh+e)
        float hiv = hh ? own2k1 : recv;    // k element e+8
        pf.h[0][e] = f2bf(lo);
        pf.h[1][e] = f2bf(hiv);
      }
#pragma unroll
      for (int td = 0; td < 4; ++td) {
        FragAB vf;
        int row = td * 16 + lr;
        vf.h[0] = *(const bf16x8*)&Vt[row][32 * ks + 16 * hh];
        vf.h[1] = *(const bf16x8*)&Vt[row][32 * ks + 16 * hh + 8];
        o[td] = __builtin_amdgcn_wmma_f32_16x16x32_bf16(
            false, vf.v, false, pf.v, (short)0, o[td], false, false);
      }
    }
    __syncthreads();
  }

  // --- finalize: O^T / l, write bf16 [b][n][h*64+d] for the output GEMM ---
  float inv = 1.0f / l_run;
  bf16* dst = AO + (size_t)(b * SEQ + q16 + lr) * INNER + h * DIM_HEAD + 8 * hh;
#pragma unroll
  for (int td = 0; td < 4; ++td) {
    bf16x8 ov;
#pragma unroll
    for (int e = 0; e < 8; ++e) ov[e] = f2bf(o[td][e] * inv);
    *(bf16x8*)(dst + 16 * td) = ov;
  }
}

// ---------------------------------------------------------------------------
extern "C" void kernel_launch(void* const* d_in, const int* in_sizes, int n_in,
                              void* d_out, int out_size, void* d_ws, size_t ws_size,
                              hipStream_t stream)
{
  (void)in_sizes; (void)n_in; (void)out_size; (void)ws_size;
  const float* x   = (const float*)d_in[0];
  const float* ctx = (const float*)d_in[1];
  const float* Wq  = (const float*)d_in[2];
  const float* Wk  = (const float*)d_in[3];
  const float* Wv  = (const float*)d_in[4];
  const float* Wo  = (const float*)d_in[5];
  const float* bo  = (const float*)d_in[6];
  float* out = (float*)d_out;

  bf16* qb = (bf16*)d_ws;                            // Q  bf16 [B*SEQ][INNER]
  bf16* k0 = qb + (size_t)BATCH * SEQ * INNER;       // k0 bf16 [B][INNER]
  bf16* v0 = k0 + BATCH * INNER;                     // v0 bf16 [B][INNER]
  bf16* ao = v0 + BATCH * INNER;                     // attn-out bf16 [B*SEQ][INNER]

  ctxproj_kernel<<<dim3(32), dim3(256), 0, stream>>>(ctx, Wk, Wv, k0, v0);

  gemm128_kernel<false, true><<<dim3(INNER / 128, BATCH * SEQ / 128), dim3(256),
                                0, stream>>>(x, Wq, qb, nullptr,
                                             BATCH * SEQ, QDIM, INNER);

  attn128_kernel<<<dim3(SEQ / 128, HEADS, BATCH), dim3(256), 0, stream>>>(
      qb, k0, v0, ao);

  gemm128_kernel<true, false><<<dim3(QDIM / 128, BATCH * SEQ / 128), dim3(256),
                                0, stream>>>(ao, Wo, out, bo,
                                             BATCH * SEQ, INNER, QDIM);
}
